// DynamicsWorldModel_48619029791151
// MI455X (gfx1250) — matrix operand order repaired and checked by reading, not compiled, not run on hardware
//
#include <hip/hip_runtime.h>
#include <hip/hip_bf16.h>
#include <math.h>

typedef __attribute__((ext_vector_type(16))) _Float16 v16h;
typedef __attribute__((ext_vector_type(8)))  _Float16 v8h;
typedef __attribute__((ext_vector_type(8)))  float    v8f;

#define BN   2
#define NH   16
#define NKV  4
#define SEQ  2048
#define DH   64
#define QSCALE 0.125f           // d^-0.5, folded into Q at conversion time
#define SC   50.0f              // softclamp
#define SPECIAL_START 2040      // 2048 - 8 special tokens

union F16x16 { v16h v; v8h h[2]; };

__device__ __forceinline__ v16h mk_frag(const _Float16* p0, const _Float16* p1) {
  F16x16 f;
  f.h[0] = *(const v8h*)p0;
  f.h[1] = *(const v8h*)p1;
  return f.v;
}

// ---- wave32 16-lane butterfly exchange, pure-VALU via DPP16 -----------------
// quad_perm xor1 = 0xB1, quad_perm xor2 = 0x4E, row_half_mirror = 0x141,
// row_mirror = 0x140 : together reduce each 16-lane half of the wave.
template <int CTRL>
__device__ __forceinline__ float dpp_xch(float x) {
  return __builtin_bit_cast(
      float, __builtin_amdgcn_mov_dpp(__builtin_bit_cast(int, x), CTRL, 0xf,
                                      0xf, true));
}
__device__ __forceinline__ float red16_max(float x) {
  x = fmaxf(x, dpp_xch<0xB1>(x));
  x = fmaxf(x, dpp_xch<0x4E>(x));
  x = fmaxf(x, dpp_xch<0x141>(x));
  x = fmaxf(x, dpp_xch<0x140>(x));
  return x;
}
__device__ __forceinline__ float red16_sum(float x) {
  x += dpp_xch<0xB1>(x);
  x += dpp_xch<0x4E>(x);
  x += dpp_xch<0x141>(x);
  x += dpp_xch<0x140>(x);
  return x;
}

// branch-free softclamp: SC * tanh(s/SC), tanh(x) = (e^2x - 1)/(e^2x + 1)
__device__ __forceinline__ float softclamp(float s) {
  const float e = __expf(s * (2.0f / SC));          // v_exp_f32
  return SC * (e - 1.0f) * __builtin_amdgcn_rcpf(e + 1.0f);
}

// ---------------- prep: RoPE + f16 convert (Q scaled, K unscaled) -------------
__global__ __launch_bounds__(256) void rope_f16_kernel(
    const float* __restrict__ x, const float* __restrict__ rot,
    _Float16* __restrict__ xo, float mul) {
  const int tid = blockIdx.x * 256 + threadIdx.x;
  const int dd  = tid & 31;
  const int pos = (tid >> 5) & (SEQ - 1);
  const int bh  = tid >> 16;
  const size_t base = ((size_t)bh * SEQ + pos) * DH;
  const float x1 = x[base + dd];
  const float x2 = x[base + dd + 32];
  const float r1 = rot[pos * DH + dd];
  const float r2 = rot[pos * DH + dd + 32];
  // rotate_half: out[dd] = x1*cos - x2*sin ; out[dd+32] = x2*cos + x1*sin
  const float o1 = x1 * __cosf(r1) - x2 * __sinf(r1);
  const float o2 = x2 * __cosf(r2) + x1 * __sinf(r2);
  xo[base + dd]      = (_Float16)(o1 * mul);
  xo[base + dd + 32] = (_Float16)(o2 * mul);
}

// ---------------- prep: V -> f16, d-major transpose Vt[b,kv][d][s] -----------
__global__ __launch_bounds__(256) void transpose_v_kernel(
    const float* __restrict__ v, _Float16* __restrict__ vt) {
  const int tid = blockIdx.x * 256 + threadIdx.x;
  const int dd  = tid & (DH - 1);
  const int pos = (tid >> 6) & (SEQ - 1);
  const int bkv = tid >> 17;
  vt[((size_t)bkv * DH + dd) * SEQ + pos] =
      (_Float16)v[((size_t)bkv * SEQ + pos) * DH + dd];
}

// ---------------- flash attention: one wave per 16-row query tile ------------
__global__ __launch_bounds__(128) void fattn_kernel(
    const _Float16* __restrict__ Qh, const _Float16* __restrict__ Kh,
    const _Float16* __restrict__ Vt, float* __restrict__ out) {
  __shared__ _Float16 Plds[4][16][32];     // per-wave P tile (16 q rows x 32 keys)

  const int lane = threadIdx.x & 31;
  const int wv   = threadIdx.x >> 5;
  const int w    = blockIdx.x * 4 + wv;    // 4096 waves total
  const int qt   = w & 127;                // query tile (16 rows each)
  const int h    = (w >> 7) & 15;          // query head
  const int b    = w >> 11;                // batch
  const int kvh  = h >> 2;                 // GQA: 4 q-heads per kv head
  const int qb   = qt << 4;                // first query row
  const int m    = lane & 15;
  const int h2   = lane >> 4;

  const _Float16* qptr = Qh + (((size_t)(b * NH + h)) * SEQ + qb) * DH;
  const _Float16* kptr = Kh + ((size_t)(b * NKV + kvh)) * SEQ * DH;
  const _Float16* vptr = Vt + ((size_t)(b * NKV + kvh)) * DH * SEQ;

  // Q A-fragments (16x32 f16): lane row m; halves {8*h2..+7, 8*h2+16..+23}
  const v16h aq0 = mk_frag(qptr + m * DH + 8 * h2,
                           qptr + m * DH + 8 * h2 + 16);
  const v16h aq1 = mk_frag(qptr + m * DH + 32 + 8 * h2,
                           qptr + m * DH + 32 + 8 * h2 + 16);

  v8f o0 = {}, o1 = {}, o2 = {}, o3 = {};  // out accum, d tiles 0..3 (C layout)
  float mrow[8], lrow[8];
#pragma unroll
  for (int r = 0; r < 8; ++r) { mrow[r] = -__builtin_inff(); lrow[r] = 0.f; }

  // block-causal: 16 | 256, so masking is purely a loop bound
  const int jmax = ((qb >> 8) + 1) << 8;

  for (int j0 = 0; j0 < jmax; j0 += 32) {
    // prefetch next key/value tile while this step computes
    if (j0 + 32 < jmax) {
      __builtin_prefetch(kptr + (size_t)(j0 + 32 + m) * DH, 0, 1);
      __builtin_prefetch(kptr + (size_t)(j0 + 48 + m) * DH, 0, 1);
      __builtin_prefetch(vptr + (size_t)m * SEQ + j0 + 32, 0, 1);
    }

    // ---- S = Qrope . Krope^T (scale pre-folded into Q) ----
    v8f s0 = {}, s1 = {};
    {
      const _Float16* pk = kptr + (size_t)(j0 + m) * DH + 8 * h2;
      v16h bk = mk_frag(pk, pk + 16);                // d 0..31
      s0 = __builtin_amdgcn_wmma_f32_16x16x32_f16(false, aq0, false, bk,
                                                  (short)0, s0, false, false);
      bk = mk_frag(pk + 32, pk + 48);                // d 32..63
      s0 = __builtin_amdgcn_wmma_f32_16x16x32_f16(false, aq1, false, bk,
                                                  (short)0, s0, false, false);
    }
    {
      const _Float16* pk = kptr + (size_t)(j0 + 16 + m) * DH + 8 * h2;
      v16h bk = mk_frag(pk, pk + 16);
      s1 = __builtin_amdgcn_wmma_f32_16x16x32_f16(false, aq0, false, bk,
                                                  (short)0, s1, false, false);
      bk = mk_frag(pk + 32, pk + 48);
      s1 = __builtin_amdgcn_wmma_f32_16x16x32_f16(false, aq1, false, bk,
                                                  (short)0, s1, false, false);
    }

    // ---- softclamp + special mask + online softmax (C/D layout) ----
    const bool tailmask = (j0 + 32) > SPECIAL_START;   // only j0 == 2016
    float alpha[8];
#pragma unroll
    for (int r = 0; r < 8; ++r) {
      float a0 = softclamp(s0[r]);
      float a1 = softclamp(s1[r]);
      if (tailmask) {
        const int row  = qb + r + 8 * h2;
        const int col1 = j0 + 16 + m;                 // sub-tile 1 key index
        if (col1 >= SPECIAL_START && row < SPECIAL_START) a1 = -3.4028235e38f;
      }
      const float mp = red16_max(fmaxf(a0, a1));
      const float mn = fmaxf(mrow[r], mp);
      const float al = __expf(mrow[r] - mn);
      const float p0 = __expf(a0 - mn);
      const float p1 = __expf(a1 - mn);
      const float sp = red16_sum(p0 + p1);
      lrow[r]  = al * lrow[r] + sp;
      mrow[r]  = mn;
      alpha[r] = al;
      // C-layout -> LDS row-major P tile (f16), same wave reads it back
      Plds[wv][r + 8 * h2][m]      = (_Float16)p0;
      Plds[wv][r + 8 * h2][m + 16] = (_Float16)p1;
    }

    // ---- P A-fragment from LDS (in-wave LDS ops are in-order) ----
    const _Float16* pp = &Plds[wv][m][8 * h2];
    const v16h ap = mk_frag(pp, pp + 16);

    // rescale running output, then accumulate P.V
#pragma unroll
    for (int r = 0; r < 8; ++r) {
      o0[r] *= alpha[r]; o1[r] *= alpha[r];
      o2[r] *= alpha[r]; o3[r] *= alpha[r];
    }
    {
      const _Float16* pv = vptr + (size_t)m * SEQ + j0 + 8 * h2;
      v16h bv = mk_frag(pv, pv + 16);
      o0 = __builtin_amdgcn_wmma_f32_16x16x32_f16(false, ap, false, bv,
                                                  (short)0, o0, false, false);
      pv += (size_t)16 * SEQ;
      bv = mk_frag(pv, pv + 16);
      o1 = __builtin_amdgcn_wmma_f32_16x16x32_f16(false, ap, false, bv,
                                                  (short)0, o1, false, false);
      pv += (size_t)16 * SEQ;
      bv = mk_frag(pv, pv + 16);
      o2 = __builtin_amdgcn_wmma_f32_16x16x32_f16(false, ap, false, bv,
                                                  (short)0, o2, false, false);
      pv += (size_t)16 * SEQ;
      bv = mk_frag(pv, pv + 16);
      o3 = __builtin_amdgcn_wmma_f32_16x16x32_f16(false, ap, false, bv,
                                                  (short)0, o3, false, false);
    }
  }

  // ---- epilogue: divide by softmax denominator, store fp32 ----
  const size_t outbase = ((size_t)(b * NH + h) * SEQ + qb) * DH;
#pragma unroll
  for (int r = 0; r < 8; ++r) {
    const float inv = __builtin_amdgcn_rcpf(lrow[r]);
    const size_t ro = outbase + (size_t)(r + 8 * h2) * DH + m;
    out[ro]      = o0[r] * inv;
    out[ro + 16] = o1[r] * inv;
    out[ro + 32] = o2[r] * inv;
    out[ro + 48] = o3[r] * inv;
  }
}

extern "C" void kernel_launch(void* const* d_in, const int* in_sizes, int n_in,
                              void* d_out, int out_size, void* d_ws, size_t ws_size,
                              hipStream_t stream) {
  const float* q   = (const float*)d_in[0];
  const float* k   = (const float*)d_in[1];
  const float* v   = (const float*)d_in[2];
  const float* rot = (const float*)d_in[3];
  float* out = (float*)d_out;

  // workspace layout (f16): Qh (8 MB) | Kh (2 MB) | Vt (2 MB)
  _Float16* Qh = (_Float16*)d_ws;
  _Float16* Kh = Qh + (size_t)BN * NH  * SEQ * DH;
  _Float16* Vt = Kh + (size_t)BN * NKV * SEQ * DH;

  // RoPE + f16 convert: Q (scaled by d^-0.5) and K
  rope_f16_kernel<<<(BN * NH  * SEQ * 32) / 256, 256, 0, stream>>>(q, rot, Qh, QSCALE);
  rope_f16_kernel<<<(BN * NKV * SEQ * 32) / 256, 256, 0, stream>>>(k, rot, Kh, 1.0f);
  // V -> f16, transposed to [d][s] for contiguous WMMA B fragments
  transpose_v_kernel<<<(BN * NKV * SEQ * DH) / 256, 256, 0, stream>>>(v, Vt);

  // flash attention: 4096 waves (one per 16-row q tile), 4 waves per block
  fattn_kernel<<<1024, 128, 0, stream>>>(Qh, Kh, Vt, out);
}
